// SSMDiffusionLM_8572754723185
// MI455X (gfx1250) — compile-verified, matrix operand order checked by reading, not run on hardware
//
#include <hip/hip_runtime.h>
#include <stdint.h>

// ---------------- problem constants ----------------
#define EMBED   1024
#define DINNER  2048
#define DSTATE  16
#define DTRANK  64
#define DCONV   4
#define BATCH   4
#define LPRIME  1024
#define NTOK    (BATCH*LPRIME)     // 4096 tokens

typedef __attribute__((ext_vector_type(16))) _Float16 v16h;
typedef __attribute__((ext_vector_type(8)))  float    v8f;

union FragH { uint4 q[2]; v16h v; };

__device__ __forceinline__ float sigm_(float x){ return 1.f/(1.f + __expf(-x)); }

// CDNA5 async global->LDS copy (ASYNCcnt-tracked), per-lane 16B.
// vdst = LDS byte address (VGPR), vaddr = 64-bit global address (VGPR pair).
#define ASYNC_B128(lds32, gptr) \
  asm volatile("global_load_async_to_lds_b128 %0, %1, off" \
               :: "v"(lds32), "v"(gptr) : "memory")
#define WAIT_ASYNC(n) asm volatile("s_wait_asynccnt " #n ::: "memory")

// ---------------- f32 -> f16 convert ----------------
__global__ __launch_bounds__(256) void cvt_f16(const float* __restrict__ s,
                                               _Float16* __restrict__ d, int n){
  int i = blockIdx.x*256 + threadIdx.x;
  if (i < n) d[i] = (_Float16)s[i];
}

// ---------------- layernorm (one block per token) ----------------
__global__ __launch_bounds__(256) void layernorm_f16(const float* __restrict__ x,
                                                     const float* __restrict__ g,
                                                     const float* __restrict__ b,
                                                     _Float16* __restrict__ out){
  __shared__ float red0[256];
  __shared__ float red1[256];
  int tok = blockIdx.x;
  const float* xr = x + (size_t)tok*EMBED;
  float v[4]; float s = 0.f, ss = 0.f;
  #pragma unroll
  for (int j=0;j<4;j++){ float t = xr[threadIdx.x + j*256]; v[j]=t; s+=t; ss+=t*t; }
  red0[threadIdx.x]=s; red1[threadIdx.x]=ss; __syncthreads();
  for (int off=128; off>0; off>>=1){
    if (threadIdx.x < off){ red0[threadIdx.x]+=red0[threadIdx.x+off];
                            red1[threadIdx.x]+=red1[threadIdx.x+off]; }
    __syncthreads();
  }
  float mu  = red0[0]*(1.f/EMBED);
  float var = red1[0]*(1.f/EMBED) - mu*mu;
  float rs  = rsqrtf(var + 1e-5f);
  #pragma unroll
  for (int j=0;j<4;j++){
    int c = threadIdx.x + j*256;
    out[(size_t)tok*EMBED + c] = (_Float16)((v[j]-mu)*rs*g[c] + b[c]);
  }
}

// ---------------- WMMA f16 GEMM:  out[M,N] = epi( A[M,K] @ W[N,K]^T ) ----------------
// MODE 0: f32 store                       MODE 1: f32 store, +bias[n]
// MODE 2: f16 store, + tc[batch(row)][n]  MODE 3: f16 store, silu(acc+bias[n])
// block = 256 thr = 8 wave32; tile 128(M) x 128(N) x 32(K); per-wave 32x64 (2x4 WMMA)
// LDS double-buffered, filled by global_load_async_to_lds_b128 (ASYNCcnt).
#define TILE_A_ELEMS (128*32)
#define TILE_B_ELEMS (128*32)
#define BUF_ELEMS    (TILE_A_ELEMS + TILE_B_ELEMS)

template<int MODE>
__global__ __launch_bounds__(256) void gemm_f16_wmma(
    const _Float16* __restrict__ A, const _Float16* __restrict__ W,
    float* __restrict__ outF, _Float16* __restrict__ outH,
    const float* __restrict__ bias, int M, int N, int K)
{
  __shared__ __align__(16) _Float16 smem[2][BUF_ELEMS];  // [buf][A(128x32) | B(128x32)]
  const int tid  = threadIdx.x;
  const int lane = tid & 31, wave = tid >> 5;            // wave32
  const int waveM = wave >> 1, waveN = wave & 1;         // 4 x 2 wave grid
  const int hi = lane >> 4,  lr = lane & 15;
  const int blockN = blockIdx.x * 128;
  const int blockM = blockIdx.y * 128;

  // staging: each tile = 128 rows x 4 b128 chunks = 512 chunks; 2 per thread (A and B alike)
  const int idx0 = tid*2,      row0 = idx0 >> 2, col0 = (idx0 & 3)*8;
  const int idx1 = tid*2 + 1,  row1 = idx1 >> 2, col1 = (idx1 & 3)*8;
  const _Float16* Ap0 = A + (size_t)(blockM + row0)*K + col0;
  const _Float16* Ap1 = A + (size_t)(blockM + row1)*K + col1;
  const _Float16* Bp0 = W + (size_t)(blockN + row0)*K + col0;
  const _Float16* Bp1 = W + (size_t)(blockN + row1)*K + col1;

  const uint32_t BUFBYTES = BUF_ELEMS * 2u;
  const uint32_t lA0 = (uint32_t)(uintptr_t)&smem[0][row0*32 + col0];
  const uint32_t lA1 = (uint32_t)(uintptr_t)&smem[0][row1*32 + col1];
  const uint32_t lB0 = (uint32_t)(uintptr_t)&smem[0][TILE_A_ELEMS + row0*32 + col0];
  const uint32_t lB1 = (uint32_t)(uintptr_t)&smem[0][TILE_A_ELEMS + row1*32 + col1];

  #define ISSUE_TILE(k0, bs) do { uint32_t bo = (uint32_t)(bs)*BUFBYTES; \
      ASYNC_B128(lA0 + bo, Ap0 + (k0)); \
      ASYNC_B128(lA1 + bo, Ap1 + (k0)); \
      ASYNC_B128(lB0 + bo, Bp0 + (k0)); \
      ASYNC_B128(lB1 + bo, Bp1 + (k0)); } while(0)

  v8f acc[2][4] = {};

  ISSUE_TILE(0, 0);
  int cur = 0;
  for (int k0 = 0; k0 < K; k0 += 32){
    if (k0 + 32 < K){ ISSUE_TILE(k0 + 32, cur ^ 1); WAIT_ASYNC(4); }
    else            { WAIT_ASYNC(0); }
    __syncthreads();                       // whole tile resident for all waves

    const _Float16* sAc = &smem[cur][0];
    const _Float16* sBc = sAc + TILE_A_ELEMS;
    // fragment gather per 16-bit A/B lane layout:
    // lane<16: K 0..7 (q0) & 16..23 (q1); lane>=16: K 8..15 & 24..31
    FragH fa[2], fb[4];
    #pragma unroll
    for (int mt=0; mt<2; mt++){
      int m = waveM*32 + mt*16 + lr;
      fa[mt].q[0] = *(const uint4*)&sAc[m*32 +      hi*8];
      fa[mt].q[1] = *(const uint4*)&sAc[m*32 + 16 + hi*8];
    }
    #pragma unroll
    for (int nt=0; nt<4; nt++){
      int n = waveN*64 + nt*16 + lr;
      fb[nt].q[0] = *(const uint4*)&sBc[n*32 +      hi*8];
      fb[nt].q[1] = *(const uint4*)&sBc[n*32 + 16 + hi*8];
    }
    #pragma unroll
    for (int mt=0; mt<2; mt++)
      #pragma unroll
      for (int nt=0; nt<4; nt++)
        acc[mt][nt] = __builtin_amdgcn_wmma_f32_16x16x32_f16(
            false, fa[mt].v, false, fb[nt].v, (short)0, acc[mt][nt], false, false);

    __syncthreads();                       // done reading before buffer reuse
    cur ^= 1;
  }
  #undef ISSUE_TILE

  #pragma unroll
  for (int mt=0; mt<2; mt++){
    #pragma unroll
    for (int nt=0; nt<4; nt++){
      #pragma unroll
      for (int r=0; r<8; r++){
        int gm = blockM + waveM*32 + mt*16 + hi*8 + r;   // D layout: VGPR r -> M = hi*8+r
        int gn = blockN + waveN*64 + nt*16 + lr;
        float vv = acc[mt][nt][r];
        size_t o = (size_t)gm*N + gn;
        if      (MODE == 0) outF[o] = vv;
        else if (MODE == 1) outF[o] = vv + bias[gn];
        else if (MODE == 2) outH[o] = (_Float16)(vv + bias[(gm >> 10)*N + gn]);
        else { float sv = vv + bias[gn]; outH[o] = (_Float16)(sv * sigm_(sv)); }
      }
    }
  }
}

// ---------------- depthwise causal conv (DCONV=4) + SiLU ----------------
__global__ __launch_bounds__(256) void conv_silu(const float* __restrict__ xz,
                                                 const float* __restrict__ cw,
                                                 const float* __restrict__ cb,
                                                 float* __restrict__ xact){
  int gid = blockIdx.x*256 + threadIdx.x;        // NTOK*DINNER
  int d   = gid & (DINNER-1);
  int tok = gid >> 11;
  int l   = tok & (LPRIME-1);
  int bb  = tok - l;                             // b*LPRIME
  float s = cb[d];
  #pragma unroll
  for (int j=0;j<DCONV;j++){
    int li = l - (DCONV-1) + j;
    if (li >= 0) s += cw[d*DCONV + j] * xz[(size_t)(bb + li)*(2*DINNER) + d];
  }
  xact[gid] = s * sigm_(s);
}

// ---------------- x-projection (96 outputs/token) + softplus on dt ----------------
__global__ __launch_bounds__(256) void xproj_k(const float* __restrict__ xact,
                                               const float* __restrict__ Wx,
                                               float* __restrict__ dt_sp,
                                               float* __restrict__ Bm,
                                               float* __restrict__ Cm){
  __shared__ float xs[DINNER];
  int tok = blockIdx.x;
  const float* xr = xact + (size_t)tok*DINNER;
  for (int i=threadIdx.x; i<DINNER; i+=256) xs[i] = xr[i];
  __syncthreads();
  int j = threadIdx.x;
  if (j < DTRANK + 2*DSTATE){
    const float* wr = Wx + (size_t)j*DINNER;
    float acc = 0.f;
    for (int k=0;k<DINNER;k++) acc = fmaf(xs[k], wr[k], acc);
    if (j < DTRANK){
      dt_sp[(size_t)tok*DTRANK + j] = (acc > 20.f) ? acc : log1pf(__expf(acc));
    } else if (j < DTRANK + DSTATE){
      Bm[(size_t)tok*DSTATE + (j-DTRANK)] = acc;
    } else {
      Cm[(size_t)tok*DSTATE + (j-DTRANK-DSTATE)] = acc;
    }
  }
}

// ---------------- selective scan: one lane per (b,d,n) state ----------------
// 131072 lanes (4096 wave32). y reduced over the 16-lane state group via shfl_xor;
// lane n==0 of each group applies D-residual + silu(z) gating and stores f16.
__global__ __launch_bounds__(256) void scan_gate(const float* __restrict__ dt_sp,
    const float* __restrict__ Bm, const float* __restrict__ Cm,
    const float* __restrict__ xact, const float* __restrict__ xz,
    const float* __restrict__ A_log, const float* __restrict__ Dp,
    _Float16* __restrict__ gated, float* __restrict__ h_out){
  int gid = blockIdx.x*256 + threadIdx.x;        // BATCH*DINNER*DSTATE = 131072
  int n  = gid & (DSTATE-1);
  int bd = gid >> 4;                              // 0..8191
  int d  = bd & (DINNER-1);
  int b  = bd >> 11;
  int r  = d >> 5;                                // repeat factor DINNER/DTRANK = 32
  float Aval  = -__expf(A_log[d*DSTATE + n]);
  float dcoef = Dp[d];
  float h = 0.f;
  for (int l=0; l<LPRIME; l++){
    int tok = (b << 10) | l;
    float dt = dt_sp[(size_t)tok*DTRANK + r];
    float x  = xact [(size_t)tok*DINNER + d];
    float Bv = Bm[(size_t)tok*DSTATE + n];
    float Cv = Cm[(size_t)tok*DSTATE + n];
    h = __expf(dt*Aval)*h + (dt*x)*Bv;
    float y = h * Cv;
    y += __shfl_xor(y, 1, 32);
    y += __shfl_xor(y, 2, 32);
    y += __shfl_xor(y, 4, 32);
    y += __shfl_xor(y, 8, 32);
    if (n == 0){
      float x_in = xz[(size_t)tok*(2*DINNER) + d];
      float z    = xz[(size_t)tok*(2*DINNER) + DINNER + d];
      float gv = (y + x_in*dcoef) * z * sigm_(z);
      gated[(size_t)tok*DINNER + d] = (_Float16)gv;
    }
  }
  h_out[(size_t)bd*DSTATE + n] = h;               // h_last, fully coalesced
}

// ---------------- sinusoidal time embedding ----------------
__global__ __launch_bounds__(256) void time_embed_k(const int* __restrict__ t,
                                                    float* __restrict__ emb){
  int gid = blockIdx.x*256 + threadIdx.x;        // BATCH*EMBED
  int b = gid >> 10, i = gid & 1023;
  const int half = EMBED/2;
  float tv  = (float)t[b];
  int idx   = (i < half) ? i : (i - half);
  float f   = __expf(-((float)idx/(float)half) * logf(10000.f));
  float ang = tv * f;
  emb[gid]  = (i < half) ? __sinf(ang) : __cosf(ang);
}

// ---------------- tiny 4-row dense layer (time MLP) ----------------
__global__ __launch_bounds__(256) void mlp_small(const float* __restrict__ in,
    const float* __restrict__ W, const float* __restrict__ bias,
    float* __restrict__ out, int applySilu){
  int gid = blockIdx.x*256 + threadIdx.x;        // BATCH*EMBED
  int b = gid >> 10, j = gid & 1023;
  const float* ir = in + (size_t)b*EMBED;
  const float* wr = W  + (size_t)j*EMBED;
  float acc = bias[j];
  for (int k=0;k<EMBED;k++) acc = fmaf(ir[k], wr[k], acc);
  if (applySilu) acc = acc * sigm_(acc);
  out[gid] = acc;
}

// =====================================================================
extern "C" void kernel_launch(void* const* d_in, const int* in_sizes, int n_in,
                              void* d_out, int out_size, void* d_ws, size_t ws_size,
                              hipStream_t stream) {
  const float* x_t   = (const float*)d_in[0];
  const int*   t     = (const int*)  d_in[1];
  const float* ln_g  = (const float*)d_in[2];
  const float* ln_b  = (const float*)d_in[3];
  const float* W_in  = (const float*)d_in[4];
  const float* convw = (const float*)d_in[5];
  const float* convb = (const float*)d_in[6];
  const float* Wxp   = (const float*)d_in[7];
  const float* A_log = (const float*)d_in[8];
  const float* Dp    = (const float*)d_in[9];
  const float* W_out = (const float*)d_in[10];
  const float* tmW1  = (const float*)d_in[11];
  const float* tmb1  = (const float*)d_in[12];
  const float* tmW2  = (const float*)d_in[13];
  const float* tmb2  = (const float*)d_in[14];
  const float* dnW1  = (const float*)d_in[15];
  const float* dnb1  = (const float*)d_in[16];
  const float* dnW2  = (const float*)d_in[17];
  const float* dnb2  = (const float*)d_in[18];

  // ---- workspace layout (bump allocator, 256B aligned) ----
  char* w = (char*)d_ws;
  size_t off = 0;
  #define WS_ALLOC(ptr_t, name, bytes) \
      ptr_t name = (ptr_t)(w + off); off = (off + (size_t)(bytes) + 255) & ~(size_t)255;

  WS_ALLOC(_Float16*, Win_h , (size_t)2*DINNER*EMBED*2)            // 4096x1024 f16
  WS_ALLOC(_Float16*, Wout_h, (size_t)EMBED*DINNER*2)              // 1024x2048
  WS_ALLOC(_Float16*, dnW1_h, (size_t)4*EMBED*EMBED*2)             // 4096x1024
  WS_ALLOC(_Float16*, dnW2_h, (size_t)EMBED*4*EMBED*2)             // 1024x4096
  WS_ALLOC(_Float16*, xn_h  , (size_t)NTOK*EMBED*2)                // 4096x1024
  WS_ALLOC(float*,    xz    , (size_t)NTOK*2*DINNER*4)             // 4096x4096 f32
  WS_ALLOC(float*,    xact  , (size_t)NTOK*DINNER*4)               // 4096x2048 f32
  WS_ALLOC(float*,    dtsp  , (size_t)NTOK*DTRANK*4)
  WS_ALLOC(float*,    Bmb   , (size_t)NTOK*DSTATE*4)
  WS_ALLOC(float*,    Cmb   , (size_t)NTOK*DSTATE*4)
  WS_ALLOC(_Float16*, gat_h , (size_t)NTOK*DINNER*2)               // 4096x2048 f16
  WS_ALLOC(_Float16*, comb_h, (size_t)NTOK*EMBED*2)                // 4096x1024 f16
  WS_ALLOC(_Float16*, h1_h  , (size_t)NTOK*4*EMBED*2)              // 4096x4096 f16
  WS_ALLOC(float*,    emb   , (size_t)BATCH*EMBED*4)
  WS_ALLOC(float*,    hmid  , (size_t)BATCH*EMBED*4)
  WS_ALLOC(float*,    tcond , (size_t)BATCH*EMBED*4)
  (void)ws_size; (void)n_in; (void)in_sizes; (void)out_size;

  float* predOut  = (float*)d_out;                                 // 4096x1024
  float* hlastOut = (float*)d_out + (size_t)NTOK*EMBED;            // 4x2048x16

  // 1) weights -> f16
  { int n = 2*DINNER*EMBED; cvt_f16<<<n/256, 256, 0, stream>>>(W_in,  Win_h,  n); }
  { int n = EMBED*DINNER;   cvt_f16<<<n/256, 256, 0, stream>>>(W_out, Wout_h, n); }
  { int n = 4*EMBED*EMBED;  cvt_f16<<<n/256, 256, 0, stream>>>(dnW1,  dnW1_h, n); }
  { int n = EMBED*4*EMBED;  cvt_f16<<<n/256, 256, 0, stream>>>(dnW2,  dnW2_h, n); }

  // 2) layernorm -> x_n (f16)
  layernorm_f16<<<NTOK, 256, 0, stream>>>(x_t, ln_g, ln_b, xn_h);

  // 3) xz = x_n @ W_in^T                       (M=4096, N=4096, K=1024) -> f32
  gemm_f16_wmma<0><<<dim3(4096/128, 4096/128), 256, 0, stream>>>(
      xn_h, Win_h, xz, nullptr, nullptr, NTOK, 2*DINNER, EMBED);

  // 4) depthwise causal conv + SiLU -> x_act
  conv_silu<<<(NTOK*DINNER)/256, 256, 0, stream>>>(xz, convw, convb, xact);

  // 5) x-proj -> dt(softplus), B, C
  xproj_k<<<NTOK, 256, 0, stream>>>(xact, Wxp, dtsp, Bmb, Cmb);

  // 6) time conditioning (tiny)
  time_embed_k<<<(BATCH*EMBED)/256, 256, 0, stream>>>(t, emb);
  mlp_small<<<(BATCH*EMBED)/256, 256, 0, stream>>>(emb,  tmW1, tmb1, hmid,  1);
  mlp_small<<<(BATCH*EMBED)/256, 256, 0, stream>>>(hmid, tmW2, tmb2, tcond, 0);

  // 7) selective scan + D-residual + silu(z) gating -> gated (f16), h_last -> d_out
  scan_gate<<<(BATCH*DINNER*DSTATE)/256, 256, 0, stream>>>(
      dtsp, Bmb, Cmb, xact, xz, A_log, Dp, gat_h, hlastOut);

  // 8) combined = gated @ W_out^T + time_cond[batch]   (M=4096, N=1024, K=2048) -> f16
  gemm_f16_wmma<2><<<dim3(1024/128, 4096/128), 256, 0, stream>>>(
      gat_h, Wout_h, nullptr, comb_h, tcond, NTOK, EMBED, DINNER);

  // 9) h1 = silu(combined @ dn_W1^T + dn_b1)           (M=4096, N=4096, K=1024) -> f16
  gemm_f16_wmma<3><<<dim3(4096/128, 4096/128), 256, 0, stream>>>(
      comb_h, dnW1_h, nullptr, h1_h, dnb1, NTOK, 4*EMBED, EMBED);

  // 10) pred_x0 = h1 @ dn_W2^T + dn_b2                 (M=4096, N=1024, K=4096) -> f32 d_out
  gemm_f16_wmma<1><<<dim3(1024/128, 4096/128), 256, 0, stream>>>(
      h1_h, dnW2_h, predOut, nullptr, dnb2, NTOK, EMBED, 4*EMBED);
}
